// GAT_66907000537774
// MI455X (gfx1250) — matrix operand order repaired and checked by reading, not compiled, not run on hardware
//
#include <hip/hip_runtime.h>

// GAT (3-layer) for MI455X gfx1250.
// GEMMs via v_wmma_f32_16x16x32_f16 (fp16 in, fp32 acc); graph side fp32 atomics.

#define NN 100000
#define NE 1000000
#define HEADS 4
#define HIDF 32
#define HHC 128
#define OUTC 40
#define NEG_SLOPE 0.2f

typedef __attribute__((ext_vector_type(16))) _Float16 v16h;
typedef __attribute__((ext_vector_type(8)))  _Float16 v8h;
typedef __attribute__((ext_vector_type(8)))  float    v8f;

static __host__ __device__ inline int cdiv(int a, int b) { return (a + b - 1) / b; }

// ---------------- f32 -> f16 convert ----------------
__global__ void cvt_f16_kernel(const float* __restrict__ in, _Float16* __restrict__ out, int n) {
    int i = blockIdx.x * blockDim.x + threadIdx.x;
    if (i < n) out[i] = (_Float16)in[i];
}

// ---------------- pack W into B-fragment order ----------------
// Output index = ((ct*(K/32)+kt)*32 + lane)*16 + i
// lane l: col = ct*16 + (l&15); k = kt*32 + (l>>4)*8 + (i&7) + (i>>3)*16  (ISA 7.12.2 B layout)
__global__ void pack_w_kernel(const float* __restrict__ W, _Float16* __restrict__ out,
                              int K, int M, int CT) {
    int idx = blockIdx.x * blockDim.x + threadIdx.x;
    int total = CT * (K / 32) * 512;
    if (idx >= total) return;
    int i    = idx & 15;
    int lane = (idx >> 4) & 31;
    int rest = idx >> 9;
    int kt   = rest % (K / 32);
    int ct   = rest / (K / 32);
    int col  = ct * 16 + (lane & 15);
    int k    = kt * 32 + (lane >> 4) * 8 + (i & 7) + (i >> 3) * 16;
    float v  = (col < M) ? W[(size_t)k * M + col] : 0.0f;
    out[idx] = (_Float16)v;
}

// ---------------- WMMA GEMM: H[N,M] = X16[N,K] @ Wpacked ----------------
template <int K, int CT>
__global__ __launch_bounds__(CT * 32) void gemm_wmma_kernel(
    const _Float16* __restrict__ X, const _Float16* __restrict__ Bp,
    float* __restrict__ H, int M) {
    const int wave = threadIdx.x >> 5;     // col tile
    const int lane = threadIdx.x & 31;
    const int rowTile = blockIdx.x;
    const int row   = rowTile * 16 + (lane & 15);
    const int kbase = (lane >> 4) * 8;     // A fragment K base (ISA 7.12.2)
    const _Float16* xrow = X + (size_t)row * K;

    v8f acc = {};
#pragma unroll
    for (int kt = 0; kt < K / 32; ++kt) {
        union { v16h v; v8h h[2]; } A;
        A.h[0] = *(const v8h*)(xrow + kt * 32 + kbase);
        A.h[1] = *(const v8h*)(xrow + kt * 32 + kbase + 16);
        const v16h b = *(const v16h*)(Bp + (((size_t)wave * (K / 32) + kt) * 32 + lane) * 16);
        acc = __builtin_amdgcn_wmma_f32_16x16x32_f16(
            false, A.v, false, b, (short)0, acc, false, false);
    }
    const int col = wave * 16 + (lane & 15);
    const int r0  = rowTile * 16 + (lane >> 4) * 8;
    if (col < M) {
#pragma unroll
        for (int i = 0; i < 8; ++i)
            H[(size_t)(r0 + i) * M + col] = acc[i];
    }
}

// ---------------- el/er = sum over F of h * al/ar ----------------
__global__ void attn_scores_kernel(const float* __restrict__ H,
                                   const float* __restrict__ al, const float* __restrict__ ar,
                                   float* __restrict__ el, float* __restrict__ er,
                                   int N, int heads, int F) {
    int i = blockIdx.x * blockDim.x + threadIdx.x;
    if (i >= N * heads) return;
    int n = i / heads, h = i % heads;
    const float* hp  = H + ((size_t)n * heads + h) * F;
    const float* alp = al + (size_t)h * F;
    const float* arp = ar + (size_t)h * F;
    float sl = 0.f, sr = 0.f;
    for (int f = 0; f < F; ++f) { float v = hp[f]; sl += v * alp[f]; sr += v * arp[f]; }
    el[i] = sl; er[i] = sr;
}

// order-preserving key for float atomicMax on unsigned
__device__ inline unsigned fkey(float f) {
    unsigned u = __float_as_uint(f);
    return (u & 0x80000000u) ? ~u : (u | 0x80000000u);
}

__global__ void edge_max_kernel(const float* __restrict__ el, const float* __restrict__ er,
                                const int* __restrict__ src, const int* __restrict__ dst,
                                unsigned* __restrict__ m, int E, int heads) {
    int i = blockIdx.x * blockDim.x + threadIdx.x;
    if (i >= E * heads) return;
    int e = i / heads, h = i % heads;
    int sn = src[e], dn = dst[e];
    float v = el[sn * heads + h] + er[dn * heads + h];
    v = v > 0.f ? v : v * NEG_SLOPE;
    atomicMax(&m[dn * heads + h], fkey(v));
}

// decode keys in place -> float; key==0 means no in-edges -> m=0 (matches isfinite guard)
__global__ void decode_m_kernel(unsigned* __restrict__ m, int n) {
    int i = blockIdx.x * blockDim.x + threadIdx.x;
    if (i >= n) return;
    unsigned k = m[i];
    float f;
    if (k == 0u) f = 0.0f;
    else if (k & 0x80000000u) f = __uint_as_float(k & 0x7FFFFFFFu);
    else f = __uint_as_float(~k);
    ((float*)m)[i] = f;
}

__global__ void edge_sum_kernel(const float* __restrict__ el, const float* __restrict__ er,
                                const int* __restrict__ src, const int* __restrict__ dst,
                                const float* __restrict__ m, float* __restrict__ s,
                                int E, int heads) {
    int i = blockIdx.x * blockDim.x + threadIdx.x;
    if (i >= E * heads) return;
    int e = i / heads, h = i % heads;
    int sn = src[e], dn = dst[e];
    float v = el[sn * heads + h] + er[dn * heads + h];
    v = v > 0.f ? v : v * NEG_SLOPE;
    atomicAdd(&s[dn * heads + h], __expf(v - m[dn * heads + h]));
}

// aggregation, 4 heads x 32 feats: one wave per edge, lane handles j = c*32+lane (head c)
__global__ __launch_bounds__(256) void edge_aggr4_kernel(
    const float* __restrict__ H, const float* __restrict__ el, const float* __restrict__ er,
    const float* __restrict__ m, const float* __restrict__ s,
    const int* __restrict__ src, const int* __restrict__ dst,
    float* __restrict__ O, int E) {
    int widx = (blockIdx.x * blockDim.x + threadIdx.x) >> 5;
    int lane = threadIdx.x & 31;
    if (widx >= E) return;
    int sn = src[widx], dn = dst[widx];
    float alpha[4];
#pragma unroll
    for (int h = 0; h < 4; ++h) {
        float e = el[sn * 4 + h] + er[dn * 4 + h];
        e = e > 0.f ? e : e * NEG_SLOPE;
        float ex = __expf(e - m[dn * 4 + h]);
        float ss = s[dn * 4 + h];
        alpha[h] = ex / (ss == 0.f ? 1.f : ss);
    }
    const float* hs = H + (size_t)sn * 128;
    float* od = O + (size_t)dn * 128;
#pragma unroll
    for (int c = 0; c < 4; ++c) {
        int j = c * 32 + lane;
        atomicAdd(&od[j], alpha[c] * hs[j]);
    }
}

// aggregation, 1 head x 40 feats: one wave per edge
__global__ __launch_bounds__(256) void edge_aggr1_kernel(
    const float* __restrict__ H, const float* __restrict__ el, const float* __restrict__ er,
    const float* __restrict__ m, const float* __restrict__ s,
    const int* __restrict__ src, const int* __restrict__ dst,
    float* __restrict__ O, int E) {
    int widx = (blockIdx.x * blockDim.x + threadIdx.x) >> 5;
    int lane = threadIdx.x & 31;
    if (widx >= E) return;
    int sn = src[widx], dn = dst[widx];
    float e = el[sn] + er[dn];
    e = e > 0.f ? e : e * NEG_SLOPE;
    float ex = __expf(e - m[dn]);
    float ss = s[dn];
    float alpha = ex / (ss == 0.f ? 1.f : ss);
    const float* hs = H + (size_t)sn * 40;
    float* od = O + (size_t)dn * 40;
    for (int j = lane; j < 40; j += 32) atomicAdd(&od[j], alpha * hs[j]);
}

// bias + relu + convert to f16 for next layer
__global__ void finish12_kernel(const float* __restrict__ O, const float* __restrict__ b,
                                _Float16* __restrict__ X16, int N) {
    int i = blockIdx.x * blockDim.x + threadIdx.x;
    if (i >= N * 128) return;
    int j = i & 127;
    float v = O[i] + b[j];
    v = v > 0.f ? v : 0.f;
    X16[i] = (_Float16)v;
}

// bias + log_softmax over 40 classes
__global__ void finish3_kernel(const float* __restrict__ O, const float* __restrict__ b,
                               float* __restrict__ out, int N) {
    int n = blockIdx.x * blockDim.x + threadIdx.x;
    if (n >= N) return;
    float v[40];
    float mx = -3.0e38f;
#pragma unroll
    for (int j = 0; j < 40; ++j) { v[j] = O[(size_t)n * 40 + j] + b[j]; mx = fmaxf(mx, v[j]); }
    float ssum = 0.f;
#pragma unroll
    for (int j = 0; j < 40; ++j) ssum += __expf(v[j] - mx);
    float ls = __logf(ssum) + mx;
#pragma unroll
    for (int j = 0; j < 40; ++j) out[(size_t)n * 40 + j] = v[j] - ls;
}

extern "C" void kernel_launch(void* const* d_in, const int* in_sizes, int n_in,
                              void* d_out, int out_size, void* d_ws, size_t ws_size,
                              hipStream_t stream) {
    const float* feat = (const float*)d_in[0];
    const float* W1 = (const float*)d_in[1];
    const float* al1 = (const float*)d_in[2];
    const float* ar1 = (const float*)d_in[3];
    const float* b1 = (const float*)d_in[4];
    const float* W2 = (const float*)d_in[5];
    const float* al2 = (const float*)d_in[6];
    const float* ar2 = (const float*)d_in[7];
    const float* b2 = (const float*)d_in[8];
    const float* W3 = (const float*)d_in[9];
    const float* al3 = (const float*)d_in[10];
    const float* ar3 = (const float*)d_in[11];
    const float* b3 = (const float*)d_in[12];
    const int* src = (const int*)d_in[13];
    const int* dst = (const int*)d_in[14];
    float* out = (float*)d_out;

    // workspace layout (all offsets 256B aligned); ~160.2 MB total
    char* ws = (char*)d_ws;
    float*    H   = (float*)(ws);                              // N*128 f32
    float*    O   = (float*)(ws + 51200000);                   // N*128 f32
    _Float16* X16 = (_Float16*)(ws + 102400000);               // N*256 f16
    float*    EL  = (float*)(ws + 153600000);                  // N*4 f32
    float*    ER  = (float*)(ws + 155200000);                  // N*4 f32
    unsigned* MB  = (unsigned*)(ws + 156800000);               // N*4 keys/floats
    float*    S   = (float*)(ws + 158400000);                  // N*4 f32
    _Float16* WP  = (_Float16*)(ws + 160000000);               // packed W, <=128KB

    const int T = 256;

    // ---------------- Layer 1: K=256, M=128, heads=4 ----------------
    cvt_f16_kernel<<<cdiv(NN * 256, T), T, 0, stream>>>(feat, X16, NN * 256);
    pack_w_kernel<<<cdiv(8 * 8 * 512, T), T, 0, stream>>>(W1, WP, 256, 128, 8);
    gemm_wmma_kernel<256, 8><<<NN / 16, 256, 0, stream>>>(X16, WP, H, 128);
    attn_scores_kernel<<<cdiv(NN * 4, T), T, 0, stream>>>(H, al1, ar1, EL, ER, NN, 4, 32);
    hipMemsetAsync(MB, 0, (size_t)NN * 4 * 4, stream);
    hipMemsetAsync(S, 0, (size_t)NN * 4 * 4, stream);
    hipMemsetAsync(O, 0, (size_t)NN * 128 * 4, stream);
    edge_max_kernel<<<cdiv(NE * 4, T), T, 0, stream>>>(EL, ER, src, dst, MB, NE, 4);
    decode_m_kernel<<<cdiv(NN * 4, T), T, 0, stream>>>(MB, NN * 4);
    edge_sum_kernel<<<cdiv(NE * 4, T), T, 0, stream>>>(EL, ER, src, dst, (float*)MB, S, NE, 4);
    edge_aggr4_kernel<<<cdiv(NE * 32, T), T, 0, stream>>>(H, EL, ER, (float*)MB, S, src, dst, O, NE);
    finish12_kernel<<<cdiv(NN * 128, T), T, 0, stream>>>(O, b1, X16, NN);

    // ---------------- Layer 2: K=128, M=128, heads=4 ----------------
    pack_w_kernel<<<cdiv(8 * 4 * 512, T), T, 0, stream>>>(W2, WP, 128, 128, 8);
    gemm_wmma_kernel<128, 8><<<NN / 16, 256, 0, stream>>>(X16, WP, H, 128);
    attn_scores_kernel<<<cdiv(NN * 4, T), T, 0, stream>>>(H, al2, ar2, EL, ER, NN, 4, 32);
    hipMemsetAsync(MB, 0, (size_t)NN * 4 * 4, stream);
    hipMemsetAsync(S, 0, (size_t)NN * 4 * 4, stream);
    hipMemsetAsync(O, 0, (size_t)NN * 128 * 4, stream);
    edge_max_kernel<<<cdiv(NE * 4, T), T, 0, stream>>>(EL, ER, src, dst, MB, NE, 4);
    decode_m_kernel<<<cdiv(NN * 4, T), T, 0, stream>>>(MB, NN * 4);
    edge_sum_kernel<<<cdiv(NE * 4, T), T, 0, stream>>>(EL, ER, src, dst, (float*)MB, S, NE, 4);
    edge_aggr4_kernel<<<cdiv(NE * 32, T), T, 0, stream>>>(H, EL, ER, (float*)MB, S, src, dst, O, NE);
    finish12_kernel<<<cdiv(NN * 128, T), T, 0, stream>>>(O, b2, X16, NN);

    // ---------------- Layer 3: K=128, M=40 (pad 48), heads=1 ----------------
    pack_w_kernel<<<cdiv(3 * 4 * 512, T), T, 0, stream>>>(W3, WP, 128, 40, 3);
    gemm_wmma_kernel<128, 3><<<NN / 16, 96, 0, stream>>>(X16, WP, H, 40);
    attn_scores_kernel<<<cdiv(NN, T), T, 0, stream>>>(H, al3, ar3, EL, ER, NN, 1, 40);
    hipMemsetAsync(MB, 0, (size_t)NN * 4, stream);
    hipMemsetAsync(S, 0, (size_t)NN * 4, stream);
    hipMemsetAsync(O, 0, (size_t)NN * 40 * 4, stream);
    edge_max_kernel<<<cdiv(NE, T), T, 0, stream>>>(EL, ER, src, dst, MB, NE, 1);
    decode_m_kernel<<<cdiv(NN, T), T, 0, stream>>>(MB, NN);
    edge_sum_kernel<<<cdiv(NE, T), T, 0, stream>>>(EL, ER, src, dst, (float*)MB, S, NE, 1);
    edge_aggr1_kernel<<<cdiv(NE * 32, T), T, 0, stream>>>(H, EL, ER, (float*)MB, S, src, dst, O, NE);
    finish3_kernel<<<cdiv(NN, 128), 128, 0, stream>>>(O, b3, out, NN);
}
